// CipherMamba_29875792511381
// MI455X (gfx1250) — matrix operand order repaired and checked by reading, compiled, not run on hardware
//
#include <hip/hip_runtime.h>
#include <hip/hip_bf16.h>

// ---------------------------------------------------------------------------
// Model dims (from the reference)
#define NL 4
#define H  512
#define IN_DIM 64
#define OUT_DIM 8
#define DI 1024          // 2*H
#define DS 16
#define DC 4
#define BB 2
#define SS 1024
#define TT (BB * SS)     // 2048 tokens
#define XPN (DI + 2 * DS)  // 1056
#define EPS 1e-5f

typedef __attribute__((ext_vector_type(16))) __bf16 v16bf;
typedef __attribute__((ext_vector_type(8)))  __bf16 v8bf;
typedef __attribute__((ext_vector_type(8)))  float  v8f;

// ---------------------------------------------------------------------------
__device__ __forceinline__ float dev_softplus(float x) {
    return (x > 20.0f) ? x : log1pf(__expf(x));
}
__device__ __forceinline__ float dev_silu(float x) {
    return x / (1.0f + __expf(-x));
}
__device__ __forceinline__ float dev_sigmoid(float x) {
    return 1.0f / (1.0f + __expf(-x));
}

// ---------------------------------------------------------------------------
// WMMA GEMM: C[M,N] = act( A[M,K](lda) * B[K,N](ldb) + bias + addend )
// f32 in memory, bf16 v_wmma_f32_16x16x32_bf16 with f32 accumulation.
// Block: 256 threads (8 waves). Tile BM=64, BN=128, BK=32, double-buffered LDS.
// Wave w -> m-tile (w&3), n-tiles (w>>2)*4 .. +3  => 4 WMMAs/wave/k-step, shared A frag.
//
// LDS layouts are fragment-native so each lane's 16-bf16 fragment is one
// contiguous 32-byte region (vector ds_load_b128 pairs, no scalar u16 gathers):
//   As[row][k]: per-row contiguous K order [0..31]
//               (lanes<16 read bytes 0..31 = K 0..15; lanes>=16 bytes 32..63 = K 16..31)
//   Bt[col][k]: B stored transposed; lane reads K 0..15 (lanes<16) / 16..31 contiguously.
//
// Double buffering: stage tile k+1 while WMMAs consume tile k; one barrier per
// k-step, so next-tile global_loads overlap the WMMA issue stream.
//
// Preconditions used here: M % 64 == 0, K % 32 == 0, N % 8 == 0 (all true for
// this model: M=2048; K in {64,512,1024}; N in {512,1024,1056,2048}).
#define BM 64
#define BN 128
#define BK 32
#define ACT_NONE 0
#define ACT_SOFTPLUS 1

__device__ __forceinline__ void gemm_stage(
    const float* __restrict__ A, int lda,
    const float* __restrict__ B, int ldb,
    int K, int block_m, int k0,
    int ar_st, int ac_st, int apos,
    int bc_st, int br_st, int gn, bool nok,
    __bf16 (* __restrict__ As)[BK],   // [BM][BK], contiguous K per row
    __bf16 (* __restrict__ Bt)[BK])   // [BN][BK], transposed
{
    // ---- stage A: 8 floats -> 8 bf16, one 16B LDS store
    {
        const float* ap = A + (block_m + ar_st) * lda + k0 + ac_st;
        float4 a0 = *(const float4*)ap;
        float4 a1 = *(const float4*)(ap + 4);
        union { __bf16 h[8]; v8bf v; } pa;
        pa.h[0] = (__bf16)a0.x; pa.h[1] = (__bf16)a0.y;
        pa.h[2] = (__bf16)a0.z; pa.h[3] = (__bf16)a0.w;
        pa.h[4] = (__bf16)a1.x; pa.h[5] = (__bf16)a1.y;
        pa.h[6] = (__bf16)a1.z; pa.h[7] = (__bf16)a1.w;
        *(v8bf*)&As[ar_st][apos] = pa.v;
        if (k0 + 2 * BK < K) __builtin_prefetch(ap + 2 * BK, 0, 1);
    }
    // ---- stage B: 2 rows x 8 cols, packed (k,k+1) pairs -> b32 LDS stores
    {
        float r0[8] = {0,0,0,0,0,0,0,0};
        float r1[8] = {0,0,0,0,0,0,0,0};
        if (nok) {
            const float* bp0 = B + (k0 + br_st) * ldb + gn;
            const float* bp1 = bp0 + ldb;
            float4 b00 = *(const float4*)bp0;
            float4 b01 = *(const float4*)(bp0 + 4);
            float4 b10 = *(const float4*)bp1;
            float4 b11 = *(const float4*)(bp1 + 4);
            r0[0]=b00.x; r0[1]=b00.y; r0[2]=b00.z; r0[3]=b00.w;
            r0[4]=b01.x; r0[5]=b01.y; r0[6]=b01.z; r0[7]=b01.w;
            r1[0]=b10.x; r1[1]=b10.y; r1[2]=b10.z; r1[3]=b10.w;
            r1[4]=b11.x; r1[5]=b11.y; r1[6]=b11.z; r1[7]=b11.w;
            if (k0 + 2 * BK < K) __builtin_prefetch(bp0 + (long)(2 * BK) * ldb, 0, 1);
        }
        #pragma unroll
        for (int j = 0; j < 8; ++j) {
            union { __bf16 h[2]; unsigned u; } p;
            p.h[0] = (__bf16)r0[j];
            p.h[1] = (__bf16)r1[j];
            *(unsigned*)&Bt[bc_st + j][br_st] = p.u;
        }
    }
}

__global__ __launch_bounds__(256) void gemm_bf16_wmma(
    const float* __restrict__ A, int lda,
    const float* __restrict__ B, int ldb,
    const float* __restrict__ bias,     // [N] or nullptr
    const float* __restrict__ addend,   // [M,ldc] or nullptr (residual)
    float* __restrict__ C, int ldc,
    int M, int N, int K, int act)
{
    __shared__ __bf16 As[2][BM][BK];   // 8 KB, contiguous K per row, ping-pong
    __shared__ __bf16 Bt[2][BN][BK];   // 16 KB, transposed, ping-pong

    const int tid  = threadIdx.x;
    const int wave = tid >> 5;
    const int lane = tid & 31;
    const int block_m = blockIdx.y * BM;
    const int block_n = blockIdx.x * BN;

    const int wm = wave & 3;            // m-tile index within block (0..3)
    const int wn = (wave >> 2) * 4;     // first n-tile index (0 or 4)

    v8f acc0 = {}, acc1 = {}, acc2 = {}, acc3 = {};

    // --- staging indices (uniform per thread across k-steps)
    const int ar_st = tid >> 2;               // 0..63
    const int ac_st = (tid & 3) * 8;          // 0,8,16,24
    // dest for an aligned 8-chunk: identity (contiguous K order)
    const int apos  = ac_st;
    const int bc_st = (tid & 15) * 8;         // 0..120
    const int br_st = (tid >> 4) * 2;         // 0..30
    const int gn    = block_n + bc_st;
    const bool nok  = (gn < N);               // chunk-granular (N % 8 == 0)

    // --- fragment indices
    const int ar   = wm * 16 + (lane & 15);
    const int aoff = (lane < 16) ? 0 : 16;
    const int bcol = lane & 15;
    const int kr   = (lane < 16) ? 0 : 16;

    // prologue: stage first tile into buffer 0
    gemm_stage(A, lda, B, ldb, K, block_m, 0,
               ar_st, ac_st, apos, bc_st, br_st, gn, nok, As[0], Bt[0]);
    __syncthreads();

    int buf = 0;
    for (int k0 = 0; k0 < K; k0 += BK, buf ^= 1) {
        // stage next tile into the other buffer (overlaps with WMMAs below)
        if (k0 + BK < K)
            gemm_stage(A, lda, B, ldb, K, block_m, k0 + BK,
                       ar_st, ac_st, apos, bc_st, br_st, gn, nok,
                       As[buf ^ 1], Bt[buf ^ 1]);

        // fragments from current buffer: contiguous 32B per lane
        v16bf afrag = *(const v16bf*)&As[buf][ar][aoff];
        v16bf bf0 = *(const v16bf*)&Bt[buf][(wn + 0) * 16 + bcol][kr];
        v16bf bf1 = *(const v16bf*)&Bt[buf][(wn + 1) * 16 + bcol][kr];
        v16bf bf2 = *(const v16bf*)&Bt[buf][(wn + 2) * 16 + bcol][kr];
        v16bf bf3 = *(const v16bf*)&Bt[buf][(wn + 3) * 16 + bcol][kr];
        acc0 = __builtin_amdgcn_wmma_f32_16x16x32_bf16(
                   false, afrag, false, bf0, (short)0, acc0, false, false);
        acc1 = __builtin_amdgcn_wmma_f32_16x16x32_bf16(
                   false, afrag, false, bf1, (short)0, acc1, false, false);
        acc2 = __builtin_amdgcn_wmma_f32_16x16x32_bf16(
                   false, afrag, false, bf2, (short)0, acc2, false, false);
        acc3 = __builtin_amdgcn_wmma_f32_16x16x32_bf16(
                   false, afrag, false, bf3, (short)0, acc3, false, false);
        __syncthreads();
    }

    // ---- store. C 16x16 f32: lane<16 -> M=v, N=lane ; lane>=16 -> M=8+v, N=lane-16
    const int row_base = block_m + wm * 16 + ((lane < 16) ? 0 : 8);
    v8f accs[4] = {acc0, acc1, acc2, acc3};
    #pragma unroll
    for (int q = 0; q < 4; ++q) {
        const int col = block_n + (wn + q) * 16 + bcol;
        if (col >= N) continue;
        const float bcoef = bias ? bias[col] : 0.0f;
        #pragma unroll
        for (int v = 0; v < 8; ++v) {
            const int row = row_base + v;
            float val = accs[q][v] + bcoef;
            if (addend) val += addend[row * ldc + col];
            if (act == ACT_SOFTPLUS) val = dev_softplus(val);
            C[row * ldc + col] = val;
        }
    }
}

// ---------------------------------------------------------------------------
// LayerNorm over last dim (h_dim), one block of 256 threads per token.
__global__ __launch_bounds__(256) void layernorm_kernel(
    const float* __restrict__ in, const float* __restrict__ g,
    const float* __restrict__ b, float* __restrict__ out, int h_dim)
{
    __shared__ float red[256];
    const int t = blockIdx.x;
    const float* row = in + (long)t * h_dim;
    float* orow = out + (long)t * h_dim;

    float s = 0.0f;
    for (int i = threadIdx.x; i < h_dim; i += 256) s += row[i];
    red[threadIdx.x] = s;
    __syncthreads();
    for (int st = 128; st > 0; st >>= 1) {
        if (threadIdx.x < st) red[threadIdx.x] += red[threadIdx.x + st];
        __syncthreads();
    }
    const float mean = red[0] / (float)h_dim;
    __syncthreads();

    float vs = 0.0f;
    for (int i = threadIdx.x; i < h_dim; i += 256) {
        float d = row[i] - mean;
        vs += d * d;
    }
    red[threadIdx.x] = vs;
    __syncthreads();
    for (int st = 128; st > 0; st >>= 1) {
        if (threadIdx.x < st) red[threadIdx.x] += red[threadIdx.x + st];
        __syncthreads();
    }
    const float rstd = rsqrtf(red[0] / (float)h_dim + EPS);

    for (int i = threadIdx.x; i < h_dim; i += 256)
        orow[i] = (row[i] - mean) * rstd * g[i] + b[i];
}

// ---------------------------------------------------------------------------
// Causal depthwise conv (DC=4 taps along S) + SiLU.
// Input is the first DI columns of xz (row stride 2*DI).
__global__ __launch_bounds__(256) void conv_silu_kernel(
    const float* __restrict__ xz, const float* __restrict__ cw,
    const float* __restrict__ cb, float* __restrict__ xx)
{
    int idx = blockIdx.x * 256 + threadIdx.x;       // over BB*SS*DI
    if (idx >= TT * DI) return;
    int d = idx % DI;
    int s = (idx / DI) % SS;
    int b = idx / (DI * SS);

    float acc = cb[d];
    #pragma unroll
    for (int k = 0; k < DC; ++k) {
        int sp = s - (DC - 1) + k;
        if (sp >= 0) {
            int t = b * SS + sp;
            acc += cw[d * DC + k] * xz[(long)t * (2 * DI) + d];
        }
    }
    xx[idx] = dev_silu(acc);
}

// ---------------------------------------------------------------------------
// Selective scan: one thread per (b, d) channel, DS=16 states in registers.
// Fuses +xx*Dp and *silu(z).  Sequential over S (the only serial part).
__global__ __launch_bounds__(256) void scan_kernel(
    const float* __restrict__ delta,   // [TT, DI]
    const float* __restrict__ xx,      // [TT, DI]
    const float* __restrict__ xz,      // [TT, 2*DI]  (z = cols DI..2DI)
    const float* __restrict__ xdbl,    // [TT, XPN]   (Bm cols DI..DI+DS, Cm next DS)
    const float* __restrict__ A_log,   // [DI, DS]
    const float* __restrict__ Dp,      // [DI]
    float* __restrict__ y)             // [TT, DI]
{
    const int d = blockIdx.x * 256 + threadIdx.x;
    const int b = blockIdx.y;
    if (d >= DI) return;

    float a[DS];
    #pragma unroll
    for (int n = 0; n < DS; ++n) a[n] = -__expf(A_log[d * DS + n]);
    float st[DS];
    #pragma unroll
    for (int n = 0; n < DS; ++n) st[n] = 0.0f;
    const float dp = Dp[d];

    for (int s = 0; s < SS; ++s) {
        const int t = b * SS + s;
        const float dl = delta[(long)t * DI + d];
        const float xv = xx[(long)t * DI + d];
        const float zv = xz[(long)t * (2 * DI) + DI + d];
        const float dxu = dl * xv;
        const float* brow = xdbl + (long)t * XPN + DI;
        const float* crow = brow + DS;
        float yv = 0.0f;
        #pragma unroll
        for (int n = 0; n < DS; ++n) {
            st[n] = __expf(dl * a[n]) * st[n] + dxu * brow[n];
            yv += st[n] * crow[n];
        }
        yv += xv * dp;
        y[(long)t * DI + d] = yv * dev_silu(zv);
    }
}

// ---------------------------------------------------------------------------
// Head: out = sigmoid(u @ head_w + head_b), N=8 (too small for WMMA tiles).
__global__ __launch_bounds__(256) void head_kernel(
    const float* __restrict__ u, const float* __restrict__ w,
    const float* __restrict__ hb, float* __restrict__ out)
{
    int idx = blockIdx.x * 256 + threadIdx.x;       // over TT*OUT_DIM
    if (idx >= TT * OUT_DIM) return;
    int o = idx % OUT_DIM;
    int t = idx / OUT_DIM;
    float acc = hb[o];
    const float* row = u + (long)t * H;
    for (int k = 0; k < H; ++k) acc += row[k] * w[k * OUT_DIM + o];
    out[idx] = dev_sigmoid(acc);
}

// ---------------------------------------------------------------------------
static inline dim3 gemm_grid(int M, int N) {
    return dim3((N + BN - 1) / BN, (M + BM - 1) / BM);
}

extern "C" void kernel_launch(void* const* d_in, const int* in_sizes, int n_in,
                              void* d_out, int out_size, void* d_ws, size_t ws_size,
                              hipStream_t stream) {
    (void)in_sizes; (void)n_in; (void)out_size; (void)ws_size;

    const float* x      = (const float*)d_in[0];
    const float* emb_w  = (const float*)d_in[1];
    const float* emb_b  = (const float*)d_in[2];
    const float* ln_g   = (const float*)d_in[3];
    const float* ln_b   = (const float*)d_in[4];
    const float* in_w   = (const float*)d_in[5];
    const float* conv_w = (const float*)d_in[6];
    const float* conv_b = (const float*)d_in[7];
    const float* xp_w   = (const float*)d_in[8];
    const float* dt_w   = (const float*)d_in[9];
    const float* dt_b   = (const float*)d_in[10];
    const float* A_log  = (const float*)d_in[11];
    const float* Dp     = (const float*)d_in[12];
    const float* out_w  = (const float*)d_in[13];
    const float* fln_g  = (const float*)d_in[14];
    const float* fln_b  = (const float*)d_in[15];
    const float* head_w = (const float*)d_in[16];
    const float* head_b = (const float*)d_in[17];
    float* out = (float*)d_out;

    // workspace carve-up (floats)
    float* ws    = (float*)d_ws;
    float* h     = ws;                      ws += TT * H;        // 2048*512
    float* u     = ws;                      ws += TT * H;
    float* xz    = ws;                      ws += TT * 2 * DI;   // 2048*2048
    float* xx    = ws;                      ws += TT * DI;
    float* xdbl  = ws;                      ws += TT * XPN;
    float* delta = ws;                      ws += TT * DI;
    float* y     = ws;                      /* ws += TT * DI; */

    // --- embed: h = x @ emb_w + emb_b        (M=2048, K=64, N=512)
    gemm_bf16_wmma<<<gemm_grid(TT, H), 256, 0, stream>>>(
        x, IN_DIM, emb_w, H, emb_b, nullptr, h, H, TT, H, IN_DIM, ACT_NONE);

    for (int i = 0; i < NL; ++i) {
        // u = LN(h)
        layernorm_kernel<<<TT, 256, 0, stream>>>(h, ln_g + i * H, ln_b + i * H, u, H);
        // xz = u @ in_w[i]                    (K=512, N=2048)
        gemm_bf16_wmma<<<gemm_grid(TT, 2 * DI), 256, 0, stream>>>(
            u, H, in_w + (long)i * H * 2 * DI, 2 * DI,
            nullptr, nullptr, xz, 2 * DI, TT, 2 * DI, H, ACT_NONE);
        // xx = silu(causal_dwconv(xz[:, :DI]) + conv_b)
        conv_silu_kernel<<<(TT * DI + 255) / 256, 256, 0, stream>>>(
            xz, conv_w + (long)i * DI * DC, conv_b + (long)i * DI, xx);
        // xdbl = xx @ xp_w[i]                 (K=1024, N=1056)
        gemm_bf16_wmma<<<gemm_grid(TT, XPN), 256, 0, stream>>>(
            xx, DI, xp_w + (long)i * DI * XPN, XPN,
            nullptr, nullptr, xdbl, XPN, TT, XPN, DI, ACT_NONE);
        // delta = softplus(xdbl[:, :DI] @ dt_w[i] + dt_b[i])   (K=1024, N=1024)
        gemm_bf16_wmma<<<gemm_grid(TT, DI), 256, 0, stream>>>(
            xdbl, XPN, dt_w + (long)i * DI * DI, DI,
            dt_b + (long)i * DI, nullptr, delta, DI, TT, DI, DI, ACT_SOFTPLUS);
        // selective scan (fused D-skip + silu(z) gate) -> y
        scan_kernel<<<dim3(DI / 256, BB), 256, 0, stream>>>(
            delta, xx, xz, xdbl, A_log + (long)i * DI * DS, Dp + (long)i * DI, y);
        // h = h + y @ out_w[i]                (K=1024, N=512, residual via addend)
        gemm_bf16_wmma<<<gemm_grid(TT, H), 256, 0, stream>>>(
            y, DI, out_w + (long)i * DI * H, H,
            nullptr, h, h, H, TT, H, DI, ACT_NONE);
    }

    // final LN + sigmoid head
    layernorm_kernel<<<TT, 256, 0, stream>>>(h, fln_g, fln_b, u, H);
    head_kernel<<<(TT * OUT_DIM + 255) / 256, 256, 0, stream>>>(u, head_w, head_b, out);
}
